// HeadAttention_28200755265693
// MI455X (gfx1250) — matrix operand order repaired and checked by reading, not compiled, not run on hardware
//
#include <hip/hip_runtime.h>
#include <hip/hip_bf16.h>

// Causal single-head attention, flash-attention style, CDNA5 WMMA (gfx1250).
//   B=8, S=2048, D=1024, H=64.  All matmuls via v_wmma_f32_16x16x32_f16.
//
// Pipeline (all on `stream`):
//   1) wswizzle : Wq/Wk/Wv (f32) -> f16, pre-swizzled into WMMA B-fragment
//                 order (one 32B vector load per fragment). norm*log2(e)
//                 folded into Wq so scores come out in the log2 domain and
//                 softmax exponentials are bare v_exp_f32.
//   2) qkv_proj : x (f32, streamed once from HBM ~64MB -> roofline ~3us) ->
//                 Q,K (f16 row-major) and V (f16 B-fragment-swizzled) in ws.
//   3) attn     : flash attention. Scores computed TRANSPOSED (S^T = K*Q^T)
//                 so the P fragment for P@V is assembled in-lane from the
//                 score accumulators -- no LDS transpose needed. Waves are
//                 independent: exact causal trip count per wave, q-tiles
//                 interleaved across blocks for load balance.
//
// Workspace layout (needs 6.5 MB):
//   [0,2MB)   Qf16   row-major, pre-scaled by 0.125*log2(e)
//   [2,4MB)   Kf16   row-major
//   [4,6MB)   Vf16   swizzled: [g/32][nt][lane][e] ; element =
//                    V[row=32*(g/32)+8*(lane>>4)+(e&7)+16*(e>>3)][16*nt+(lane&15)]
//   [6MB,..)  Wsw    f16 swizzled weights, 384 KB

#define BATCH 8
#define SEQ   2048
#define EMBED 1024
#define HEAD  64

typedef __attribute__((ext_vector_type(16))) _Float16 v16h;
typedef __attribute__((ext_vector_type(8)))  _Float16 v8h;
typedef __attribute__((ext_vector_type(8)))  float    v8f;

static __device__ __forceinline__ v8f wmma_f16(v16h a, v16h b, v8f c) {
  // D = A(16x32 f16) * B(32x16 f16) + C(16x16 f32)
  return __builtin_amdgcn_wmma_f32_16x16x32_f16(
      /*neg_a=*/false, a, /*neg_b=*/false, b,
      /*c_mod=*/(short)0, c, /*reuse_a=*/false, /*reuse_b=*/false);
}

static __device__ __forceinline__ v16h cat8(v8h lo, v8h hi) {
  v16h r;
#pragma unroll
  for (int i = 0; i < 8; ++i) { r[i] = lo[i]; r[i + 8] = hi[i]; }
  return r;
}

// ---------------------------------------------------------------------------
// Kernel 0: swizzle W into B-fragment order, f32 -> f16.
// Wsw flat index t = (((o*32 + ks)*4 + nt)*32 + lane)*16 + e
// holds W_o[ks*32 + 16*(lane>>4) + e][nt*16 + (lane&15)].
// o==0 (Wq) scaled by HEAD^-0.5 * log2(e) so scores are log2-domain.
// ---------------------------------------------------------------------------
__global__ __launch_bounds__(256) void wswizzle(const float* __restrict__ Wq,
                                                const float* __restrict__ Wk,
                                                const float* __restrict__ Wv,
                                                _Float16* __restrict__ Wsw) {
  const int t    = blockIdx.x * 256 + threadIdx.x;  // exactly 196608 threads
  const int e    = t & 15;
  const int lane = (t >> 4) & 31;
  const int nt   = (t >> 9) & 3;
  const int ks   = (t >> 11) & 31;
  const int o    = t >> 16;
  const int krow = ks * 32 + ((lane >> 4) << 4) + e;
  const int col  = nt * 16 + (lane & 15);
  const float* W = (o == 0) ? Wq : (o == 1) ? Wk : Wv;
  float v = W[krow * HEAD + col];
  if (o == 0) v *= 0.125f * 1.4426950408889634f;  // norm * log2(e)
  Wsw[t] = (_Float16)v;
}

// ---------------------------------------------------------------------------
// Kernel 1: QKV projection. 4 waves/block, 16 rows/wave, 64 rows/block.
// Per wave: 12 f32 accum tiles (q,k,v) x 4 N-tiles; K-loop of 32 steps of 32.
// ---------------------------------------------------------------------------
__global__ __launch_bounds__(128) void qkv_proj(const float* __restrict__ x,
                                                const _Float16* __restrict__ Wsw,
                                                _Float16* __restrict__ qf,
                                                _Float16* __restrict__ kf,
                                                _Float16* __restrict__ vsw) {
  const int lane = threadIdx.x & 31;
  const int wave = threadIdx.x >> 5;
  const int row0 = blockIdx.x * 64 + wave * 16;  // [0, 16384)
  const int arow = lane & 15;
  const int koff = (lane >> 4) * 16;  // A-fragment K offset per lane half

  v8f acc[12];
#pragma unroll
  for (int f = 0; f < 12; ++f)
#pragma unroll
    for (int i = 0; i < 8; ++i) acc[f][i] = 0.f;

  const float* xrow = x + (size_t)(row0 + arow) * EMBED;
  const _Float16* wl = Wsw + lane * 16;

  for (int ks = 0; ks < 32; ++ks) {
    const int k = ks * 32;
    // A fragment: row = lane&15, elements 0-7 -> K=koff+i, 8-15 -> K=koff+8+i
    const v8f x0 = *(const v8f*)(xrow + k + koff);
    const v8f x1 = *(const v8f*)(xrow + k + koff + 8);
    v16h a;
#pragma unroll
    for (int i = 0; i < 8; ++i) { a[i] = (_Float16)x0[i]; a[i + 8] = (_Float16)x1[i]; }

    const _Float16* wk = wl + ks * 2048;
#pragma unroll
    for (int o = 0; o < 3; ++o)
#pragma unroll
      for (int nt = 0; nt < 4; ++nt)
        acc[o * 4 + nt] =
            wmma_f16(a, *(const v16h*)(wk + o * 65536 + nt * 512), acc[o * 4 + nt]);
  }

  // C-layout: VGPR r -> row r + 8*(lane>>4); lane&15 -> col within N-tile.
  const int crow = (lane >> 4) * 8;
  const int ccol = lane & 15;
#pragma unroll
  for (int r = 0; r < 8; ++r) {
    const int g = row0 + crow + r;  // global row = b*SEQ + s
#pragma unroll
    for (int nt = 0; nt < 4; ++nt) {
      qf[(size_t)g * HEAD + nt * 16 + ccol] = (_Float16)acc[nt][r];
      kf[(size_t)g * HEAD + nt * 16 + ccol] = (_Float16)acc[4 + nt][r];
      // V stored pre-swizzled as WMMA B-fragments for the PV matmul:
      vsw[(size_t)((g >> 5) * 4 + nt) * 512 + ((g >> 3) & 1) * 256 + ccol * 16 +
          (g & 7) + ((g >> 4) & 1) * 8] = (_Float16)acc[8 + nt][r];
    }
  }
}

// ---------------------------------------------------------------------------
// Kernel 2: flash attention. 4 independent waves/block, 16 q-rows/wave.
// q-tiles interleaved across blocks (tile = blockIdx.x + wave*256) ; all 4
// waves of a block share one q-tile (same causal span), in 4 batches; each wave runs its exact
// causal trip count. Scores computed transposed: S^T = K * Q^T (M=kv, N=q),
// so the exp'd score tiles ARE the A fragment of P for O += P @ V.
// ---------------------------------------------------------------------------
__global__ __launch_bounds__(128) void attn(const _Float16* __restrict__ qf,
                                            const _Float16* __restrict__ kf,
                                            const _Float16* __restrict__ vsw,
                                            float* __restrict__ out) {
  const int lane = threadIdx.x & 31;
  const int wave = threadIdx.x >> 5;
  const int tile = blockIdx.x + wave * 256;  // [0, 1024) wave-tiles
  const int b    = tile >> 7;                // 8 batches
  const int q0   = (tile & 127) * 16;        // 128 q-tiles of 16 rows
  const size_t rbase = (size_t)b * SEQ;

  const int bcol = lane & 15;         // B-fragment column (q row for Q^T)
  const int kb   = (lane >> 4) * 16;  // B-fragment K base per lane half
  const int koff = (lane >> 4) * 16;  // A-fragment K offset per lane half
  const int rsel = (lane >> 4) * 8;   // C-layout row offset per lane half
  const int hsel = (lane & 16) >> 1;  // 0 or 8, for per-row broadcasts

  // Q^T B-fragments (Q pre-scaled by 0.125*log2e): one 32B load each.
  const _Float16* qrow = qf + (rbase + q0 + bcol) * HEAD;
  const v16h qb0 = *(const v16h*)(qrow + kb);
  const v16h qb1 = *(const v16h*)(qrow + 32 + kb);

  v8f o0, o1, o2, o3;
#pragma unroll
  for (int i = 0; i < 8; ++i) { o0[i] = 0.f; o1[i] = 0.f; o2[i] = 0.f; o3[i] = 0.f; }
  float mi = -1e30f, li = 0.f;     // online-softmax state for q row (q0+bcol)
  const int myq = q0 + bcol;

  const int jend = q0 + 16;        // exact causal bound: kv <= q0+15
  for (int j = 0; j < jend; j += 32) {
    // ---- scores^T: two 16(kv) x 16(q) tiles, K-dim = HEAD = 64 ----
    const _Float16* kp = kf + (rbase + j + bcol) * HEAD;  // A: row = lane&15
    v8f s0, s1;
#pragma unroll
    for (int i = 0; i < 8; ++i) { s0[i] = 0.f; s1[i] = 0.f; }

    v16h ak = cat8(*(const v8h*)(kp + koff), *(const v8h*)(kp + 8 + koff));
    s0 = wmma_f16(ak, qb0, s0);
    ak = cat8(*(const v8h*)(kp + 32 + koff), *(const v8h*)(kp + 40 + koff));
    s0 = wmma_f16(ak, qb1, s0);
    const _Float16* kp1 = kp + 16 * HEAD;
    ak = cat8(*(const v8h*)(kp1 + koff), *(const v8h*)(kp1 + 8 + koff));
    s1 = wmma_f16(ak, qb0, s1);
    ak = cat8(*(const v8h*)(kp1 + 32 + koff), *(const v8h*)(kp1 + 40 + koff));
    s1 = wmma_f16(ak, qb1, s1);

    if (j + 32 < jend) __builtin_prefetch(kp + 32 * HEAD, 0, 1);

    // ---- causal mask + online softmax (rows<->lanes in S^T layout) ----
    float rm = -1e30f;
#pragma unroll
    for (int r = 0; r < 8; ++r) {
      const int kv = j + rsel + r;            // kv row of element r
      if (kv > myq)      s0[r] = -1e30f;
      if (kv + 16 > myq) s1[r] = -1e30f;
      rm = fmaxf(rm, fmaxf(s0[r], s1[r]));
    }
    rm = fmaxf(rm, __shfl_xor(rm, 16, 32));   // combine lane halves of row
    const float mn = fmaxf(mi, rm);
    float rs = 0.f;
    v16h p;  // A fragment of P for O += P @ V, assembled in-lane
#pragma unroll
    for (int r = 0; r < 8; ++r) {
      // log2-domain scores: bare v_exp_f32, no log2e multiply
      const float p0 = __builtin_amdgcn_exp2f(s0[r] - mn);
      const float p1 = __builtin_amdgcn_exp2f(s1[r] - mn);
      rs += p0 + p1;
      p[r]     = (_Float16)p0;
      p[r + 8] = (_Float16)p1;
    }
    rs += __shfl_xor(rs, 16, 32);
    const float sc = __builtin_amdgcn_exp2f(mi - mn);
    li = li * sc + rs;
    mi = mn;

    // rescale O accumulators (O rows live in VGPR index, state lives in lanes)
#pragma unroll
    for (int r = 0; r < 8; ++r) {
      const float scr = __shfl(sc, r + hsel, 16);
      o0[r] *= scr; o1[r] *= scr; o2[r] *= scr; o3[r] *= scr;
    }

    // ---- O += P @ V : V B-fragments are one 32B load each (pre-swizzled) ----
    const _Float16* vp =
        vsw + (size_t)(((b << 6) + (j >> 5)) * 4) * 512 + lane * 16;
    o0 = wmma_f16(p, *(const v16h*)(vp),        o0);
    o1 = wmma_f16(p, *(const v16h*)(vp + 512),  o1);
    o2 = wmma_f16(p, *(const v16h*)(vp + 1024), o2);
    o3 = wmma_f16(p, *(const v16h*)(vp + 1536), o3);
  }

  // ---- normalize and store f32 output ----
  const float inv = __builtin_amdgcn_rcpf(li);
#pragma unroll
  for (int r = 0; r < 8; ++r) {
    const float ivr = __shfl(inv, r + hsel, 16);
    float* op = out + (rbase + q0 + rsel + r) * HEAD;
    op[bcol]      = o0[r] * ivr;
    op[16 + bcol] = o1[r] * ivr;
    op[32 + bcol] = o2[r] * ivr;
    op[48 + bcol] = o3[r] * ivr;
  }
}

// ---------------------------------------------------------------------------
extern "C" void kernel_launch(void* const* d_in, const int* in_sizes, int n_in,
                              void* d_out, int out_size, void* d_ws, size_t ws_size,
                              hipStream_t stream) {
  (void)in_sizes; (void)n_in; (void)out_size; (void)ws_size;
  const float* x  = (const float*)d_in[0];  // embeddings [8,2048,1024]
  const float* Wq = (const float*)d_in[1];  // [1024,64]
  const float* Wk = (const float*)d_in[2];
  const float* Wv = (const float*)d_in[3];

  char* ws = (char*)d_ws;
  _Float16* qf  = (_Float16*)(ws);
  _Float16* kf  = (_Float16*)(ws + (size_t)2 * 1024 * 1024);
  _Float16* vsw = (_Float16*)(ws + (size_t)4 * 1024 * 1024);
  _Float16* Wsw = (_Float16*)(ws + (size_t)6 * 1024 * 1024);
  float* out = (float*)d_out;

  wswizzle<<<768, 256, 0, stream>>>(Wq, Wk, Wv, Wsw);         // 196608 threads
  qkv_proj<<<256, 128, 0, stream>>>(x, Wsw, qf, kf, vsw);     // 16384 rows
  attn<<<256, 128, 0, stream>>>(qf, kf, vsw, out);            // 1024 wave-tiles
}
